// MultiHeadAttention_24240795418966
// MI455X (gfx1250) — matrix operand order repaired and checked by reading, not compiled, not run on hardware
//
#include <hip/hip_runtime.h>

// ---------- types matching CDNA5 WMMA operand shapes (wave32) ----------
typedef __attribute__((ext_vector_type(16))) __bf16 v16bf;   // A/B frag: 16x32 / 32x16 bf16
typedef __attribute__((ext_vector_type(8)))  float  v8f;     // C/D frag: 16x16 f32
typedef __attribute__((ext_vector_type(4)))  unsigned int u32x4;

union Frag16 { v16bf bf; u32x4 u[2]; };

__device__ __forceinline__ v16bf load_frag_g(const __bf16* p0, const __bf16* p1) {
  Frag16 f;
  f.u[0] = *(const u32x4*)p0;
  f.u[1] = *(const u32x4*)(p1);
  return f.bf;
}

__device__ __forceinline__ v8f wmma_bf16(v16bf a, v16bf b, v8f c) {
  return __builtin_amdgcn_wmma_f32_16x16x32_bf16(false, a, false, b, (short)0, c,
                                                 false, false);
}

// Pin a fragment: loads can't sink below it, WMMAs consuming it can't hoist above it,
// and overlapping live ranges force distinct register tuples (no per-WMMA wait-0).
__device__ __forceinline__ void pin_frag(v16bf& x) { asm volatile("" : "+v"(x)); }

// CDNA5 async global->LDS copy (16B per lane, ASYNCcnt, no VGPR round-trip).
__device__ __forceinline__ void async_stage_b128(void* lds_ptr, const void* gptr) {
  unsigned lo = (unsigned)(unsigned long long)lds_ptr;   // low 32 bits = LDS offset
  asm volatile("global_load_async_to_lds_b128 %0, %1, off"
               :: "v"(lo), "v"((unsigned long long)gptr) : "memory");
}
__device__ __forceinline__ void wait_async0() {
  asm volatile("s_wait_asynccnt 0x0" ::: "memory");
}

#define DIMV   1024
#define SEQ    2048
#define NHEAD  16
#define HDIM   64
#define SCALE  0.125f
#define BSTRIDE 72                       // padded LDS row stride (bf16), 64 data + 8 pad
#define BBUF    (64 * BSTRIDE)           // one 64x64 B-tile buffer (elements)

// ---------- fp32 -> bf16 conversion ----------
__global__ void cvt_f32_bf16(const float* __restrict__ src, __bf16* __restrict__ dst, int n) {
  int i = blockIdx.x * blockDim.x + threadIdx.x;
  const int stride = gridDim.x * blockDim.x;
  for (; i < n; i += stride) dst[i] = (__bf16)src[i];
}

// ---------- shared GEMM mainloop: 8 waves x (16M x 64N), async-staged B, K-step 64 ----
__device__ __forceinline__ void gemm_mainloop(
    const __bf16* __restrict__ arow,   // A row (m0 + l15), stride DIMV, bf16
    const __bf16* __restrict__ wtile,  // B: 64 weight rows starting at n0, stride DIMV
    __bf16* __restrict__ bsh,          // LDS staging: 2 x 64 x BSTRIDE bf16
    int tid, int l15, int half, v8f acc[4])
{
  // 256 threads stage a 64(row) x 64(k) bf16 tile: 2 x b128 async per thread
  const int srow = tid >> 2, schunk = tid & 3;       // 4 x 32B chunks per row
  const __bf16* sg = wtile + (size_t)srow * DIMV + schunk * 16;
  __bf16* sl0 = bsh + srow * BSTRIDE + schunk * 16;

  async_stage_b128(sl0,     sg);                      // k-step 0
  async_stage_b128(sl0 + 8, sg + 8);
  wait_async0();
  __syncthreads();

  int cur = 0;
  for (int kk = 0; kk < DIMV; kk += 64) {
    // kick off next tile copy (overlaps with the 8 WMMAs below)
    if (kk + 64 < DIMV) {
      __bf16* d = sl0 + (cur ^ 1) * BBUF;
      async_stage_b128(d,     sg + kk + 64);
      async_stage_b128(d + 8, sg + kk + 72);
    }
    if (kk + 256 < DIMV) __builtin_prefetch(arow + kk + 256, 0, 3);

    // A fragments (direct global) + all 8 B fragments from LDS
    v16bf af0 = load_frag_g(arow + kk + 16 * half,      arow + kk + 8 + 16 * half);
    v16bf af1 = load_frag_g(arow + kk + 32 + 16 * half, arow + kk + 40 + 16 * half);
    v16bf b0[4], b1[4];
#pragma unroll
    for (int t = 0; t < 4; ++t) {
      const __bf16* bl = bsh + cur * BBUF + (t * 16 + l15) * BSTRIDE + 16 * half;
      Frag16 f;
      f.u[0] = *(const u32x4*)bl;        f.u[1] = *(const u32x4*)(bl + 8);
      b0[t] = f.bf;
      f.u[0] = *(const u32x4*)(bl + 32); f.u[1] = *(const u32x4*)(bl + 40);
      b1[t] = f.bf;
    }
    // pin: all 20 loads issue first, frags live in distinct regs, WMMAs stay below
    pin_frag(af0); pin_frag(af1);
#pragma unroll
    for (int t = 0; t < 4; ++t) { pin_frag(b0[t]); pin_frag(b1[t]); }

#pragma unroll
    for (int t = 0; t < 4; ++t) acc[t] = wmma_bf16(af0, b0[t], acc[t]);
#pragma unroll
    for (int t = 0; t < 4; ++t) acc[t] = wmma_bf16(af1, b1[t], acc[t]);

    wait_async0();                       // async copy done (overlapped with WMMAs)
    __syncthreads();
    cur ^= 1;
  }
}

// ---------- QKV GEMM: [4096,1024] x W^T -> Q (scaled) / K / V^T, head-major bf16 -----
__global__ __launch_bounds__(256) void qkv_gemm_kernel(
    const __bf16* __restrict__ xb, const __bf16* __restrict__ wb,
    const float* __restrict__ bias,
    __bf16* __restrict__ Qb, __bf16* __restrict__ Kb, __bf16* __restrict__ Vt)
{
  __shared__ __align__(16) __bf16 bsh[2 * BBUF];
  const int tid = threadIdx.x;
  const int ln = tid & 31, wid = tid >> 5;
  const int half = ln >> 4, l15 = ln & 15;
  const int m0 = blockIdx.y * 128 + wid * 16;
  const int n0 = blockIdx.x * 64;

  v8f acc[4] = {};
  gemm_mainloop(xb + (size_t)(m0 + l15) * DIMV, wb + (size_t)n0 * DIMV,
                bsh, tid, l15, half, acc);

#pragma unroll
  for (int t = 0; t < 4; ++t) {
    const int ng = n0 + t * 16 + l15;
    const float bv = bias[ng];
    const int seg = ng >> 10;                 // 0=Q 1=K 2=V (uniform per tile)
    const int c = ng & 1023;
    const int h = c >> 6, d = c & 63;
#pragma unroll
    for (int r = 0; r < 8; ++r) {
      const int mg = m0 + r + 8 * half;       // C layout: VGPR r -> row r + 8*half
      const int bb = mg >> 11, s = mg & 2047;
      float v = acc[t][r] + bv;
      if (seg == 0) {
        Qb[(((size_t)(bb * NHEAD + h) * SEQ) + s) * HDIM + d] = (__bf16)(v * SCALE);
      } else if (seg == 1) {
        Kb[(((size_t)(bb * NHEAD + h) * SEQ) + s) * HDIM + d] = (__bf16)v;
      } else {
        Vt[(((size_t)(bb * NHEAD + h) * HDIM) + d) * SEQ + s] = (__bf16)v;
      }
    }
  }
}

// ---------- Flash attention: per wave = 16 q-rows of one (b,h), online softmax -------
__global__ __launch_bounds__(128) void attn_kernel(
    const __bf16* __restrict__ Qb, const __bf16* __restrict__ Kb,
    const __bf16* __restrict__ Vt, __bf16* __restrict__ AO)
{
  __shared__ __align__(16) __bf16 pbuf[4][16][32];   // per-wave P-relayout staging

  const int ln = threadIdx.x & 31, wid = threadIdx.x >> 5;
  const int half = ln >> 4, l15 = ln & 15;
  const int b = blockIdx.z, h = blockIdx.y;
  const int q0 = blockIdx.x * 64 + wid * 16;

  const __bf16* Qh = Qb + ((size_t)(b * NHEAD + h) * SEQ) * HDIM;
  const __bf16* Kh = Kb + ((size_t)(b * NHEAD + h) * SEQ) * HDIM;
  const __bf16* Vh = Vt + ((size_t)(b * NHEAD + h) * HDIM) * SEQ;

  const __bf16* qrow = Qh + (size_t)(q0 + l15) * HDIM;
  const v16bf aq0 = load_frag_g(qrow + 16 * half,      qrow + 8 + 16 * half);
  const v16bf aq1 = load_frag_g(qrow + 32 + 16 * half, qrow + 40 + 16 * half);

  v8f accO[4] = {};
  float row_m[8], row_l[8];
#pragma unroll
  for (int r = 0; r < 8; ++r) { row_m[r] = -3.0e38f; row_l[r] = 0.0f; }

  for (int kb = 0; kb < SEQ; kb += 32) {
    // ---- issue ALL loads for this block up front (K frags then V frags) ----
    const __bf16* k0r = Kh + (size_t)(kb + l15) * HDIM + 16 * half;
    const __bf16* k1r = Kh + (size_t)(kb + 16 + l15) * HDIM + 16 * half;
    v16bf bk00 = load_frag_g(k0r,      k0r + 8);
    v16bf bk01 = load_frag_g(k0r + 32, k0r + 40);
    v16bf bk10 = load_frag_g(k1r,      k1r + 8);
    v16bf bk11 = load_frag_g(k1r + 32, k1r + 40);

    v16bf bv[4];
#pragma unroll
    for (int t = 0; t < 4; ++t) {
      const __bf16* vr = Vh + (size_t)(t * 16 + l15) * SEQ + kb + 16 * half;
      bv[t] = load_frag_g(vr, vr + 8);       // stays in flight across softmax
    }
    if (kb + 128 < SEQ) {                    // warm near cache ~4 blocks ahead
      __builtin_prefetch(k0r + 128 * HDIM, 0, 3);
      __builtin_prefetch(Vh + (size_t)l15 * SEQ + kb + 128, 0, 3);
    }
    // pin K frags only: score WMMAs wait loadcnt<=8, V loads remain outstanding
    pin_frag(bk00); pin_frag(bk01); pin_frag(bk10); pin_frag(bk11);

    // ---- S = Q K^T (scale folded into Q) ----
    v8f s0 = {}, s1 = {};
    s0 = wmma_bf16(aq0, bk00, s0);
    s0 = wmma_bf16(aq1, bk01, s0);
    s1 = wmma_bf16(aq0, bk10, s1);
    s1 = wmma_bf16(aq1, bk11, s1);

    // ---- online softmax over the 32 new keys (V loads hidden under this) ----
#pragma unroll
    for (int r = 0; r < 8; ++r) {
      float t = fmaxf(s0[r], s1[r]);
      t = fmaxf(t, __shfl_xor(t, 1, 16));
      t = fmaxf(t, __shfl_xor(t, 2, 16));
      t = fmaxf(t, __shfl_xor(t, 4, 16));
      t = fmaxf(t, __shfl_xor(t, 8, 16));
      const float mnew  = fmaxf(row_m[r], t);
      const float alpha = __expf(row_m[r] - mnew);
      const float p0 = __expf(s0[r] - mnew);
      const float p1 = __expf(s1[r] - mnew);
      float rs = p0 + p1;
      rs += __shfl_xor(rs, 1, 16);
      rs += __shfl_xor(rs, 2, 16);
      rs += __shfl_xor(rs, 4, 16);
      rs += __shfl_xor(rs, 8, 16);
      row_l[r] = row_l[r] * alpha + rs;
      row_m[r] = mnew;
#pragma unroll
      for (int t4 = 0; t4 < 4; ++t4) accO[t4][r] *= alpha;
      pbuf[wid][r + 8 * half][l15]      = (__bf16)p0;
      pbuf[wid][r + 8 * half][16 + l15] = (__bf16)p1;
    }

    // ---- reload P as 16x32 A-fragment (wave-private LDS, in-order DS) ----
    Frag16 pf;
    pf.u[0] = *(const u32x4*)&pbuf[wid][l15][16 * half];
    pf.u[1] = *(const u32x4*)&pbuf[wid][l15][8 + 16 * half];
    v16bf pa = pf.bf;

    // pin V frags here: their loads resolved only now, after the softmax window
    pin_frag(bv[0]); pin_frag(bv[1]); pin_frag(bv[2]); pin_frag(bv[3]);

    // ---- O += P V ----
#pragma unroll
    for (int t = 0; t < 4; ++t) accO[t] = wmma_bf16(pa, bv[t], accO[t]);
  }

  // ---- normalize, store [b, s, dim] bf16 for proj ----
#pragma unroll
  for (int r = 0; r < 8; ++r) {
    const float inv = 1.0f / row_l[r];
    const int sq = q0 + r + 8 * half;
#pragma unroll
    for (int t = 0; t < 4; ++t) {
      AO[((size_t)(b * SEQ + sq)) * DIMV + h * HDIM + t * 16 + l15] =
          (__bf16)(accO[t][r] * inv);
    }
  }
}

// ---------- Proj GEMM: [4096,1024] x [1024,1024] + bias -> f32 out ----------
__global__ __launch_bounds__(256) void proj_gemm_kernel(
    const __bf16* __restrict__ ab, const __bf16* __restrict__ wb,
    const float* __restrict__ bias, float* __restrict__ out)
{
  __shared__ __align__(16) __bf16 bsh[2 * BBUF];
  const int tid = threadIdx.x;
  const int ln = tid & 31, wid = tid >> 5;
  const int half = ln >> 4, l15 = ln & 15;
  const int m0 = blockIdx.y * 128 + wid * 16;
  const int n0 = blockIdx.x * 64;

  v8f acc[4] = {};
  gemm_mainloop(ab + (size_t)(m0 + l15) * DIMV, wb + (size_t)n0 * DIMV,
                bsh, tid, l15, half, acc);

#pragma unroll
  for (int t = 0; t < 4; ++t) {
    const int ng = n0 + t * 16 + l15;
    const float bv = bias[ng];
#pragma unroll
    for (int r = 0; r < 8; ++r) {
      const int mg = m0 + r + 8 * half;
      out[(size_t)mg * DIMV + ng] = acc[t][r] + bv;
    }
  }
}

// ---------- launch ----------
extern "C" void kernel_launch(void* const* d_in, const int* in_sizes, int n_in,
                              void* d_out, int out_size, void* d_ws, size_t ws_size,
                              hipStream_t stream) {
  const float* x      = (const float*)d_in[0];
  const float* qkv_w  = (const float*)d_in[1];
  const float* qkv_b  = (const float*)d_in[2];
  const float* proj_w = (const float*)d_in[3];
  const float* proj_b = (const float*)d_in[4];
  float* out = (float*)d_out;

  // Workspace layout (40 MB; AO reuses the xb region after the QKV GEMM)
  char* ws = (char*)d_ws;
  __bf16* xb   = (__bf16*)(ws);                        //  8 MB [0,8)
  __bf16* wqkv = (__bf16*)(ws + (size_t)( 8u << 20));  //  6 MB [8,14)
  __bf16* wprj = (__bf16*)(ws + (size_t)(14u << 20));  //  2 MB [14,16)
  __bf16* Qb   = (__bf16*)(ws + (size_t)(16u << 20));  //  8 MB [16,24)
  __bf16* Kb   = (__bf16*)(ws + (size_t)(24u << 20));  //  8 MB [24,32)
  __bf16* Vt   = (__bf16*)(ws + (size_t)(32u << 20));  //  8 MB [32,40)
  __bf16* AO   = (__bf16*)(ws);                        //  8 MB reuse of xb

  cvt_f32_bf16<<<512, 256, 0, stream>>>(x,      xb,   2 * SEQ * DIMV);
  cvt_f32_bf16<<<512, 256, 0, stream>>>(qkv_w,  wqkv, 3 * DIMV * DIMV);
  cvt_f32_bf16<<<256, 256, 0, stream>>>(proj_w, wprj, DIMV * DIMV);

  qkv_gemm_kernel<<<dim3(48, 32), 256, 0, stream>>>(xb, wqkv, qkv_b, Qb, Kb, Vt);
  attn_kernel<<<dim3(SEQ / 64, NHEAD, 2), 128, 0, stream>>>(Qb, Kb, Vt, AO);
  proj_gemm_kernel<<<dim3(16, 32), 256, 0, stream>>>(AO, wprj, proj_b, out);
}
